// TrialTransformer_79645873537276
// MI455X (gfx1250) — matrix operand order repaired and checked by reading, not compiled, not run on hardware
//
#include <hip/hip_runtime.h>
#include <math.h>

// ---------------- types ----------------
typedef __attribute__((ext_vector_type(16))) __bf16 v16bf_t;
typedef __attribute__((ext_vector_type(8)))  float  v8f_t;

union F16x16 { uint4 q[2]; v16bf_t v; };   // 32 bytes: one bf16 WMMA A/B fragment

__device__ __forceinline__ unsigned short f2bf(float f) {
    unsigned u = __float_as_uint(f);
    u += 0x7fffu + ((u >> 16) & 1u);       // round-to-nearest-even
    return (unsigned short)(u >> 16);
}

#define WMMA_BF16(A, B, C) \
    __builtin_amdgcn_wmma_f32_16x16x32_bf16(false, (A), false, (B), (short)0, (C), false, false)

// ---------------- elementwise converts ----------------
__global__ void cvt_bf16_kernel(const float* __restrict__ in,
                                unsigned short* __restrict__ out, int n) {
    int i = blockIdx.x * 256 + threadIdx.x;
    if (i < n) out[i] = f2bf(in[i]);
}

// W (K x N) fp32 -> Wt (N x K) bf16
__global__ void wt_kernel(const float* __restrict__ W,
                          unsigned short* __restrict__ Wt, int K, int N) {
    int i = blockIdx.x * 256 + threadIdx.x;
    if (i >= K * N) return;
    int k = i / N, n = i - k * N;
    Wt[(size_t)n * K + k] = f2bf(W[i]);
}

// ---------------- generic bf16 WMMA GEMM (128x128 block tile) ----------------
// C[M x N] = A[M x K](bf16) @ Wt[N x K](bf16)^T + bias
// 256 threads = 8 waves in a 4x2 grid; each wave computes a 32x64 sub-tile
// (2 A-frags x 4 B-frags -> 8 WMMAs per 32-deep K step).
// epi 0: QKV permute (B,L,D)->(B,H,L,DH) bf16, *scale
// epi 1: f32 out = acc + bias + residual
// epi 2: bf16 out = gelu(acc + bias)
__global__ __launch_bounds__(256)
void gemm_bf16_kernel(const unsigned short* __restrict__ A,
                      const unsigned short* __restrict__ Wt,
                      const float* __restrict__ bias,
                      float* __restrict__ outF,
                      unsigned short* __restrict__ outB,
                      const float* __restrict__ residual,
                      int M, int N, int K, int epi, float scale, int Lseq) {
    __shared__ unsigned short As[128][32];   // 8 KB
    __shared__ unsigned short Ws[128][32];   // 8 KB

    const int t = threadIdx.x;
    const int lane = t & 31, wave = t >> 5;
    const int lh = lane >> 4, ln16 = lane & 15;
    const int wm = wave >> 1, wn = wave & 1;          // 4 x 2 wave grid
    const int m0 = blockIdx.y * 128, n0 = blockIdx.x * 128;

    v8f_t acc[2][4];
#pragma unroll
    for (int i = 0; i < 2; i++)
#pragma unroll
        for (int j = 0; j < 4; j++) acc[i][j] = (v8f_t)0.0f;

    const int lrow = t >> 1;            // 0..127
    const int lkh = (t & 1) * 16;       // 0 or 16

    for (int k0 = 0; k0 < K; k0 += 32) {
        const uint4* ga = (const uint4*)(A + (size_t)(m0 + lrow) * K + k0 + lkh);
        uint4 a0 = ga[0], a1 = ga[1];
        const uint4* gw = (const uint4*)(Wt + (size_t)(n0 + lrow) * K + k0 + lkh);
        uint4 w0 = gw[0], w1 = gw[1];
        __syncthreads();
        *(uint4*)&As[lrow][lkh] = a0; *(uint4*)&As[lrow][lkh + 8] = a1;
        *(uint4*)&Ws[lrow][lkh] = w0; *(uint4*)&Ws[lrow][lkh + 8] = w1;
        __syncthreads();

        F16x16 fa[2], fb[4];
#pragma unroll
        for (int i = 0; i < 2; i++) {
            int ra = 32 * wm + 16 * i + ln16;               // A: m-row
            fa[i].q[0] = *(const uint4*)&As[ra][8 * lh];        // k = 8*lh + e
            fa[i].q[1] = *(const uint4*)&As[ra][16 + 8 * lh];   // k = 16+8*lh+e
        }
#pragma unroll
        for (int j = 0; j < 4; j++) {
            int rb = 64 * wn + 16 * j + ln16;               // B: n-row of Wt
            fb[j].q[0] = *(const uint4*)&Ws[rb][8 * lh];        // k = 8*lh + e
            fb[j].q[1] = *(const uint4*)&Ws[rb][16 + 8 * lh];
        }
#pragma unroll
        for (int i = 0; i < 2; i++)
#pragma unroll
            for (int j = 0; j < 4; j++)
                acc[i][j] = WMMA_BF16(fa[i].v, fb[j].v, acc[i][j]);
    }

#pragma unroll
    for (int i = 0; i < 2; i++)
#pragma unroll
        for (int j = 0; j < 4; j++)
#pragma unroll
            for (int r = 0; r < 8; r++) {
                int row = m0 + 32 * wm + 16 * i + r + 8 * lh;
                int col = n0 + 64 * wn + 16 * j + ln16;
                float v = acc[i][j][r] + bias[col];
                if (epi == 0) {
                    v *= scale;
                    int bb = row / Lseq, ll = row - bb * Lseq;
                    int hh = col >> 6, dh = col & 63;
                    outB[(((size_t)(bb * 8 + hh)) * Lseq + ll) * 64 + dh] = f2bf(v);
                } else if (epi == 1) {
                    size_t o = (size_t)row * N + col;
                    outF[o] = v + residual[o];
                } else {
                    float v3 = v * v * v;
                    float g = 0.5f * v * (1.0f + tanhf(0.7978845608f * (v + 0.044715f * v3)));
                    outB[(size_t)row * N + col] = f2bf(g);
                }
            }
}

// ---------------- flash attention (DH=64, bf16 WMMA, online softmax) ----------------
// Q,K,V laid out (B,H,L,64) bf16; Q pre-scaled by 1/sqrt(DH). ctx out: (B,L,512) bf16.
__global__ __launch_bounds__(128)
void attn_kernel(const unsigned short* __restrict__ Q,
                 const unsigned short* __restrict__ Kb,
                 const unsigned short* __restrict__ Vb,
                 const int* __restrict__ mask,
                 unsigned short* __restrict__ ctx, int L) {
    __shared__ unsigned short Vt[64][40];       // V chunk transposed: [dh][key], pitch 40
    __shared__ unsigned short Pst[4][16][32];   // per-wave P tile staging

    const int bh = blockIdx.x;
    const int b = bh >> 3, h = bh & 7;
    const int t = threadIdx.x;
    const int wave = t >> 5, lane = t & 31;
    const int lh = lane >> 4, ln16 = lane & 15;
    const int q0 = blockIdx.y * 64 + wave * 16;
    const size_t base = (size_t)bh * L * 64;

    // Q A-fragments (16 rows x dh 0..63), loaded once
    F16x16 aQ[2];
#pragma unroll
    for (int s = 0; s < 2; s++) {
        const unsigned short* qp = Q + base + (size_t)(q0 + ln16) * 64 + 32 * s + 8 * lh;
        aQ[s].q[0] = *(const uint4*)qp;
        aQ[s].q[1] = *(const uint4*)(qp + 16);
    }

    v8f_t accV[4];
#pragma unroll
    for (int i = 0; i < 4; i++) accV[i] = (v8f_t)0.0f;
    float mstat[8], lstat[8];
#pragma unroll
    for (int r = 0; r < 8; r++) { mstat[r] = -3.0e38f; lstat[r] = 0.0f; }

    const int vkey = t & 31;           // cooperative V staging assignment
    const int vdh = (t >> 5) * 16;

    for (int kc = 0; kc < L; kc += 32) {
        // stage V[kc..kc+31][0..63] transposed into LDS (shared by all 4 waves)
        const uint4* gv = (const uint4*)(Vb + base + (size_t)(kc + vkey) * 64 + vdh);
        uint4 v0 = gv[0], v1 = gv[1];
        unsigned short tmp[16];
        *(uint4*)tmp = v0; *(uint4*)(tmp + 8) = v1;
        __syncthreads();
#pragma unroll
        for (int i = 0; i < 16; i++) Vt[vdh + i][vkey] = tmp[i];
        __syncthreads();

        // scores: S0 = keys kc..kc+15, S1 = keys kc+16..kc+31
        v8f_t c0 = (v8f_t)0.0f, c1 = (v8f_t)0.0f;
#pragma unroll
        for (int s = 0; s < 2; s++) {
            F16x16 bk0, bk1;
            const unsigned short* kp0 = Kb + base + (size_t)(kc + ln16) * 64 + 32 * s + 8 * lh;
            bk0.q[0] = *(const uint4*)kp0; bk0.q[1] = *(const uint4*)(kp0 + 16);
            const unsigned short* kp1 = Kb + base + (size_t)(kc + 16 + ln16) * 64 + 32 * s + 8 * lh;
            bk1.q[0] = *(const uint4*)kp1; bk1.q[1] = *(const uint4*)(kp1 + 16);
            c0 = WMMA_BF16(aQ[s].v, bk0.v, c0);
            c1 = WMMA_BF16(aQ[s].v, bk1.v, c1);
        }

        // online softmax per row (row = r + 8*lh, spread over 16 lanes of a half-wave)
        float alpha[8];
#pragma unroll
        for (int r = 0; r < 8; r++) {
            int qi = q0 + r + 8 * lh;
            float b0 = (mask[(size_t)qi * L + kc + ln16] > 0) ? 0.0f : -1.0e9f;
            float b1 = (mask[(size_t)qi * L + kc + 16 + ln16] > 0) ? 0.0f : -1.0e9f;
            float s0 = c0[r] + b0, s1 = c1[r] + b1;
            float mx = fmaxf(s0, s1);
#pragma unroll
            for (int mm = 8; mm >= 1; mm >>= 1) mx = fmaxf(mx, __shfl_xor(mx, mm, 32));
            float mnew = fmaxf(mstat[r], mx);
            float al = expf(mstat[r] - mnew);
            float p0 = expf(s0 - mnew), p1 = expf(s1 - mnew);
            float rs = p0 + p1;
#pragma unroll
            for (int mm = 8; mm >= 1; mm >>= 1) rs += __shfl_xor(rs, mm, 32);
            lstat[r] = lstat[r] * al + rs;
            mstat[r] = mnew;
            alpha[r] = al;
            Pst[wave][r + 8 * lh][ln16]      = f2bf(p0);
            Pst[wave][r + 8 * lh][ln16 + 16] = f2bf(p1);
        }
#pragma unroll
        for (int tt = 0; tt < 4; tt++)
#pragma unroll
            for (int r = 0; r < 8; r++) accV[tt][r] *= alpha[r];

        // reload P as A-fragment (16x32) and do P @ V
        F16x16 aP;
        aP.q[0] = *(const uint4*)&Pst[wave][ln16][8 * lh];
        aP.q[1] = *(const uint4*)&Pst[wave][ln16][16 + 8 * lh];
#pragma unroll
        for (int tt = 0; tt < 4; tt++) {
            F16x16 bv;
            bv.q[0] = *(const uint4*)&Vt[16 * tt + ln16][8 * lh];
            bv.q[1] = *(const uint4*)&Vt[16 * tt + ln16][16 + 8 * lh];
            accV[tt] = WMMA_BF16(aP.v, bv.v, accV[tt]);
        }
    }

    // normalize and write ctx in (B, L, D) bf16 layout
#pragma unroll
    for (int tt = 0; tt < 4; tt++)
#pragma unroll
        for (int r = 0; r < 8; r++) {
            int qi = q0 + r + 8 * lh;
            float o = accV[tt][r] / lstat[r];
            ctx[((size_t)(b * L + qi)) * 512 + h * 64 + 16 * tt + ln16] = f2bf(o);
        }
}

// ---------------- LayerNorm over D=512 ----------------
__global__ __launch_bounds__(256)
void ln_kernel(const float* __restrict__ in, const float* __restrict__ g,
               const float* __restrict__ bta, float* __restrict__ outF,
               unsigned short* __restrict__ outB) {
    __shared__ float wsum[8], wsq[8];
    const int row = blockIdx.x, t = threadIdx.x;
    const size_t o = (size_t)row * 512;
    float x0 = in[o + t], x1 = in[o + 256 + t];
    float s = x0 + x1;
#pragma unroll
    for (int mm = 16; mm >= 1; mm >>= 1) s += __shfl_xor(s, mm, 32);
    if ((t & 31) == 0) wsum[t >> 5] = s;
    __syncthreads();
    float mu = 0.0f;
#pragma unroll
    for (int i = 0; i < 8; i++) mu += wsum[i];
    mu *= (1.0f / 512.0f);
    float d0 = x0 - mu, d1 = x1 - mu;
    float sq = d0 * d0 + d1 * d1;
#pragma unroll
    for (int mm = 16; mm >= 1; mm >>= 1) sq += __shfl_xor(sq, mm, 32);
    if ((t & 31) == 0) wsq[t >> 5] = sq;
    __syncthreads();
    float var = 0.0f;
#pragma unroll
    for (int i = 0; i < 8; i++) var += wsq[i];
    var *= (1.0f / 512.0f);
    float inv = rsqrtf(var + 1e-12f);
    float y0 = d0 * inv * g[t] + bta[t];
    float y1 = d1 * inv * g[256 + t] + bta[256 + t];
    outF[o + t] = y0; outF[o + 256 + t] = y1;
    if (outB) { outB[o + t] = f2bf(y0); outB[o + 256 + t] = f2bf(y1); }
}

// ---------------- host orchestration ----------------
extern "C" void kernel_launch(void* const* d_in, const int* in_sizes, int n_in,
                              void* d_out, int out_size, void* d_ws, size_t ws_size,
                              hipStream_t stream) {
    (void)in_sizes; (void)n_in; (void)out_size; (void)ws_size;
    const int B = 8, L = 1536, D = 512, FFN = 2048, M = B * L;

    const float* x_in = (const float*)d_in[0];
    const int* mask = (const int*)d_in[1 + 3 * 16];

    char* ws = (char*)d_ws;
    size_t off = 0;
    auto alloc = [&](size_t bytes) -> void* {
        void* p = ws + off;
        off = (off + bytes + 255) & ~(size_t)255;
        return p;
    };
    float* tmpF = (float*)alloc((size_t)M * D * 4);
    float* xF   = (float*)alloc((size_t)M * D * 4);
    unsigned short* xB   = (unsigned short*)alloc((size_t)M * D * 2);
    unsigned short* qB   = (unsigned short*)alloc((size_t)M * D * 2);
    unsigned short* kB   = (unsigned short*)alloc((size_t)M * D * 2);
    unsigned short* vB   = (unsigned short*)alloc((size_t)M * D * 2);
    unsigned short* ctxB = (unsigned short*)alloc((size_t)M * D * 2);
    unsigned short* hB   = (unsigned short*)alloc((size_t)M * FFN * 2);
    unsigned short* wqT  = (unsigned short*)alloc((size_t)D * D * 2);
    unsigned short* wkT  = (unsigned short*)alloc((size_t)D * D * 2);
    unsigned short* wvT  = (unsigned short*)alloc((size_t)D * D * 2);
    unsigned short* woT  = (unsigned short*)alloc((size_t)D * D * 2);
    unsigned short* w1T  = (unsigned short*)alloc((size_t)D * FFN * 2);
    unsigned short* w2T  = (unsigned short*)alloc((size_t)FFN * D * 2);

    cvt_bf16_kernel<<<(M * D + 255) / 256, 256, 0, stream>>>(x_in, xB, M * D);
    const float* cur = x_in;

    for (int layer = 0; layer < 3; layer++) {
        void* const* p = d_in + 1 + layer * 16;
        const float *wq = (const float*)p[0],  *bq = (const float*)p[1];
        const float *wk = (const float*)p[2],  *bk = (const float*)p[3];
        const float *wv = (const float*)p[4],  *bv = (const float*)p[5];
        const float *wo = (const float*)p[6],  *bo = (const float*)p[7];
        const float *l1g = (const float*)p[8], *l1b = (const float*)p[9];
        const float *w1 = (const float*)p[10], *b1 = (const float*)p[11];
        const float *w2 = (const float*)p[12], *b2 = (const float*)p[13];
        const float *l2g = (const float*)p[14],*l2b = (const float*)p[15];

        wt_kernel<<<(D * D + 255) / 256, 256, 0, stream>>>(wq, wqT, D, D);
        wt_kernel<<<(D * D + 255) / 256, 256, 0, stream>>>(wk, wkT, D, D);
        wt_kernel<<<(D * D + 255) / 256, 256, 0, stream>>>(wv, wvT, D, D);
        wt_kernel<<<(D * D + 255) / 256, 256, 0, stream>>>(wo, woT, D, D);
        wt_kernel<<<(D * FFN + 255) / 256, 256, 0, stream>>>(w1, w1T, D, FFN);
        wt_kernel<<<(FFN * D + 255) / 256, 256, 0, stream>>>(w2, w2T, FFN, D);

        dim3 gDD(D / 128, M / 128);
        // Q/K/V projections (Q pre-scaled by 1/sqrt(64))
        gemm_bf16_kernel<<<gDD, 256, 0, stream>>>(xB, wqT, bq, nullptr, qB, nullptr,
                                                  M, D, D, 0, 0.125f, L);
        gemm_bf16_kernel<<<gDD, 256, 0, stream>>>(xB, wkT, bk, nullptr, kB, nullptr,
                                                  M, D, D, 0, 1.0f, L);
        gemm_bf16_kernel<<<gDD, 256, 0, stream>>>(xB, wvT, bv, nullptr, vB, nullptr,
                                                  M, D, D, 0, 1.0f, L);

        attn_kernel<<<dim3(B * 8, L / 64), 128, 0, stream>>>(qB, kB, vB, mask, ctxB, L);

        // out projection + residual
        gemm_bf16_kernel<<<gDD, 256, 0, stream>>>(ctxB, woT, bo, tmpF, nullptr, cur,
                                                  M, D, D, 1, 1.0f, L);
        ln_kernel<<<M, 256, 0, stream>>>(tmpF, l1g, l1b, xF, xB);

        // FFN
        gemm_bf16_kernel<<<dim3(FFN / 128, M / 128), 256, 0, stream>>>(
            xB, w1T, b1, nullptr, hB, nullptr, M, FFN, D, 2, 1.0f, L);
        gemm_bf16_kernel<<<gDD, 256, 0, stream>>>(hB, w2T, b2, tmpF, nullptr, xF,
                                                  M, D, FFN, 1, 1.0f, L);
        float* lnOut = (layer == 2) ? (float*)d_out : xF;
        ln_kernel<<<M, 256, 0, stream>>>(tmpF, l2g, l2b, lnOut, xB);
        cur = xF;
    }
}
